// RNNDecoderWithHardAttention_50826642981293
// MI455X (gfx1250) — matrix operand order repaired and checked by reading, not compiled, not run on hardware
//
#include <hip/hip_runtime.h>
#include <hip/hip_bf16.h>
#include <math.h>

typedef __attribute__((ext_vector_type(16))) __bf16 v16bf;
typedef __attribute__((ext_vector_type(8)))  __bf16 v8bf;
typedef __attribute__((ext_vector_type(4)))  __bf16 v4bf;
typedef __attribute__((ext_vector_type(8)))  float  v8f;

#define BB      128
#define LL      196
#define D_ENC   2048
#define D_ATT   512
#define D_EMB   512
#define D_DEC   512
#define VOCAB   10000
#define MAX_LEN 21
#define T_STEPS 20

__device__ __forceinline__ __bf16 f2bf(float f) {
    unsigned u = __builtin_bit_cast(unsigned, f);
    u = (u + 0x7FFFu + ((u >> 16) & 1u)) >> 16;
    unsigned short s = (unsigned short)u;
    return __builtin_bit_cast(__bf16, s);
}

__device__ __forceinline__ float sigf(float x) { return 1.0f / (1.0f + __expf(-x)); }

// 8-element bf16 fragment piece loaders (16B each).
__device__ __forceinline__ v8bf load8(const __bf16* p) { return *(const v8bf*)p; }
__device__ __forceinline__ v8bf load8(const float* p) {   // fallback: inline convert
    float4 a = ((const float4*)p)[0], b = ((const float4*)p)[1];
    v8bf r;
    r[0]=f2bf(a.x); r[1]=f2bf(a.y); r[2]=f2bf(a.z); r[3]=f2bf(a.w);
    r[4]=f2bf(b.x); r[5]=f2bf(b.y); r[6]=f2bf(b.z); r[7]=f2bf(b.w);
    return r;
}

// OUT(M,N) = act( A(M,K)[lda] * W(N,K)^T + bias + add ), W pre-converted bf16.
// One wave = 64x16 tile: 4 A frags + 1 B frag -> 4 WMMAs per K-step of 32.
// Requirements: M%64==0, N%16==0, K%64==0 (all satisfied here).
template <typename AT>
__global__ __launch_bounds__(128)
void wmma_gemm_kernel(const AT* __restrict__ A, int lda,
                      const __bf16* __restrict__ W,         // (N,K) row-major bf16
                      const float* __restrict__ bias,       // len N or null
                      const float* __restrict__ addsrc, int ldadd,
                      float* __restrict__ out, long long ldo,
                      int M, int N, int K,
                      int act,                              // 0 none, 1 sigmoid
                      const int* __restrict__ lengths, int tstep) // row-mask: t < len[row]-1
{
    const int lane  = threadIdx.x & 31;
    const int wave  = threadIdx.x >> 5;
    const int nT    = N >> 4;
    const int mT    = M >> 6;
    const int tile  = blockIdx.x * 4 + wave;
    if (tile >= mT * nT) return;                 // wave-uniform: EXEC stays all-ones
    const int mt    = tile / nT;
    const int nt    = tile % nT;
    const int mBase = mt << 6;
    const int nBase = nt << 4;
    const int half  = lane >> 4;                 // 0 or 1
    const int r     = lane & 15;

    v8f acc[4] = {};

    const __bf16* wrow = W + (size_t)(nBase + r) * (size_t)K;
    const AT* ar[4];
    ar[0] = A + (size_t)(mBase      + r) * (size_t)lda;
    ar[1] = A + (size_t)(mBase + 16 + r) * (size_t)lda;
    ar[2] = A + (size_t)(mBase + 32 + r) * (size_t)lda;
    ar[3] = A + (size_t)(mBase + 48 + r) * (size_t)lda;

    #pragma unroll 2
    for (int k0 = 0; k0 < K; k0 += 32) {
        // B fragment: column (nBase+r); elems 0..7 <- K=k0+half*8.., elems 8..15 <- K=k0+16+half*8..
        v8bf blo = load8(wrow + k0 + half * 8);
        v8bf bhi = load8(wrow + k0 + 16 + half * 8);
        v16bf bf = __builtin_shufflevector(blo, bhi, 0,1,2,3,4,5,6,7,8,9,10,11,12,13,14,15);
        // A fragments: row r; elems 0..7 <- K=k0+half*8.., elems 8..15 <- K=k0+16+half*8..
        v16bf af[4];
        #pragma unroll
        for (int a = 0; a < 4; ++a) {
            v8bf lo = load8(ar[a] + k0 + half * 8);
            v8bf hi = load8(ar[a] + k0 + 16 + half * 8);
            af[a] = __builtin_shufflevector(lo, hi, 0,1,2,3,4,5,6,7,8,9,10,11,12,13,14,15);
        }
        acc[0] = __builtin_amdgcn_wmma_f32_16x16x32_bf16(false, af[0], false, bf, (short)0, acc[0], false, false);
        acc[1] = __builtin_amdgcn_wmma_f32_16x16x32_bf16(false, af[1], false, bf, (short)0, acc[1], false, false);
        acc[2] = __builtin_amdgcn_wmma_f32_16x16x32_bf16(false, af[2], false, bf, (short)0, acc[2], false, false);
        acc[3] = __builtin_amdgcn_wmma_f32_16x16x32_bf16(false, af[3], false, bf, (short)0, acc[3], false, false);
    }

    // Epilogue: lane r holds column nBase+r; VGPR v holds row mBase+a*16+half*8+v.
    const int col = nBase + r;
    const float bv = bias ? bias[col] : 0.0f;
    #pragma unroll
    for (int a = 0; a < 4; ++a) {
        #pragma unroll
        for (int v = 0; v < 8; ++v) {
            int row = mBase + a * 16 + half * 8 + v;
            float val = acc[a][v] + bv;
            if (addsrc)  val += addsrc[(size_t)row * (size_t)ldadd + col];
            if (act == 1) val = sigf(val);
            if (lengths) { if (!(tstep < lengths[row] - 1)) val = 0.0f; }
            out[(size_t)row * (size_t)ldo + col] = val;
        }
    }
}

// fp32 -> bf16 bulk conversion (4 elems/thread, 8B packed stores). n must be %4==0.
__global__ void cvt_bf16_kernel(const float* __restrict__ in, __bf16* __restrict__ out, int n4) {
    int i = blockIdx.x * blockDim.x + threadIdx.x;
    if (i >= n4) return;
    float4 f = ((const float4*)in)[i];
    v4bf o;
    o[0] = f2bf(f.x); o[1] = f2bf(f.y); o[2] = f2bf(f.z); o[3] = f2bf(f.w);
    ((v4bf*)out)[i] = o;
}

__global__ void mean_kernel(const float* __restrict__ features, __bf16* __restrict__ mean_bf) {
    int idx = blockIdx.x * blockDim.x + threadIdx.x;   // B*D_ENC threads
    int b = idx / D_ENC, d = idx % D_ENC;
    const float* p = features + (size_t)b * LL * D_ENC + d;
    float s = 0.0f;
    for (int l = 0; l < LL; ++l) s += p[(size_t)l * D_ENC];
    mean_bf[idx] = f2bf(s * (1.0f / (float)LL));
}

__global__ void hbf_init_kernel(const float* __restrict__ hc, __bf16* __restrict__ h_bf) {
    int idx = blockIdx.x * blockDim.x + threadIdx.x;   // B*D_DEC threads
    int b = idx / D_DEC, d = idx % D_DEC;
    h_bf[idx] = f2bf(hc[b * (2 * D_DEC) + d]);
}

__global__ __launch_bounds__(256)
void scores_kernel(const float* __restrict__ enc_proj, const float* __restrict__ dec_proj,
                   const float* __restrict__ w_full, const float* __restrict__ b_full,
                   float* __restrict__ scores)
{
    int lane = threadIdx.x & 31;
    int wave = threadIdx.x >> 5;
    int gid = blockIdx.x * 8 + wave;                   // = b*L + l
    if (gid >= BB * LL) return;
    int b = gid / LL;
    const float* ep = enc_proj + (size_t)gid * D_ATT;
    const float* dp = dec_proj + (size_t)b * D_ATT;
    float s = 0.0f;
    for (int a = lane; a < D_ATT; a += 32) {
        float v = ep[a] + dp[a];
        v = fmaxf(v, 0.0f);
        s += v * w_full[a];
    }
    for (int off = 16; off > 0; off >>= 1) s += __shfl_down(s, off, 32);
    if (lane == 0) scores[gid] = s + b_full[0];
}

__global__ __launch_bounds__(256)
void softmax_kernel(const float* __restrict__ scores, const float* __restrict__ temp,
                    float* __restrict__ alpha)
{
    __shared__ float sm[256];
    int b = blockIdx.x, l = threadIdx.x;
    float invT = 1.0f / temp[0];
    float x = (l < LL) ? scores[b * LL + l] * invT : -3.0e38f;
    sm[l] = x; __syncthreads();
    for (int off = 128; off > 0; off >>= 1) { if (l < off) sm[l] = fmaxf(sm[l], sm[l + off]); __syncthreads(); }
    float mx = sm[0]; __syncthreads();
    float e = (l < LL) ? __expf(x - mx) : 0.0f;
    sm[l] = e; __syncthreads();
    for (int off = 128; off > 0; off >>= 1) { if (l < off) sm[l] += sm[l + off]; __syncthreads(); }
    float denom = sm[0];
    if (l < LL) alpha[b * LL + l] = e / denom;
}

__global__ void ctx_bf_kernel(const __bf16* __restrict__ feat_bf, const float* __restrict__ alpha,
                              float* __restrict__ ctx)
{
    int idx = blockIdx.x * blockDim.x + threadIdx.x;   // B*D_ENC threads
    int b = idx / D_ENC, d = idx % D_ENC;
    const __bf16* fp = feat_bf + (size_t)b * LL * D_ENC + d;
    const float* ap = alpha + b * LL;
    float s = 0.0f;
    for (int l = 0; l < LL; ++l) s += ap[l] * (float)fp[(size_t)l * D_ENC];
    ctx[idx] = s;
}

__global__ void ctx_f32_kernel(const float* __restrict__ features, const float* __restrict__ alpha,
                               float* __restrict__ ctx)
{
    int idx = blockIdx.x * blockDim.x + threadIdx.x;   // B*D_ENC threads
    int b = idx / D_ENC, d = idx % D_ENC;
    const float* fp = features + (size_t)b * LL * D_ENC + d;
    const float* ap = alpha + b * LL;
    float s = 0.0f;
    for (int l = 0; l < LL; ++l) s += ap[l] * fp[(size_t)l * D_ENC];
    ctx[idx] = s;
}

__global__ void xasm_kernel(const int* __restrict__ captions, const float* __restrict__ embed_W,
                            const float* __restrict__ gate, const float* __restrict__ ctx,
                            __bf16* __restrict__ x_bf, int t)
{
    int idx = blockIdx.x * blockDim.x + threadIdx.x;   // B*(D_EMB+D_ENC) threads
    int b = idx / (D_EMB + D_ENC);
    int j = idx % (D_EMB + D_ENC);
    float v;
    if (j < D_EMB) {
        int tok = captions[b * MAX_LEN + t];
        v = embed_W[(size_t)tok * D_EMB + j];
    } else {
        int d = j - D_EMB;
        v = gate[b * D_ENC + d] * ctx[b * D_ENC + d];
    }
    x_bf[idx] = f2bf(v);
}

__global__ void lstm_kernel(const float* __restrict__ gates, float* __restrict__ hc,
                            __bf16* __restrict__ h_bf, __bf16* __restrict__ hn_bf,
                            const int* __restrict__ lengths, int t)
{
    int idx = blockIdx.x * blockDim.x + threadIdx.x;   // B*D_DEC threads
    int b = idx / D_DEC, d = idx % D_DEC;
    const float* g = gates + (size_t)b * (4 * D_DEC);
    float gi = g[d], gf = g[D_DEC + d], gg = g[2 * D_DEC + d], go = g[3 * D_DEC + d];
    float hold = hc[b * (2 * D_DEC) + d];
    float c    = hc[b * (2 * D_DEC) + D_DEC + d];
    float cn = sigf(gf) * c + sigf(gi) * tanhf(gg);
    float hnew = sigf(go) * tanhf(cn);
    hn_bf[idx] = f2bf(hnew);
    bool m = (t < lengths[b] - 1);
    float hsel = m ? hnew : hold;
    if (m) {
        hc[b * (2 * D_DEC) + d] = hnew;
        hc[b * (2 * D_DEC) + D_DEC + d] = cn;
    }
    h_bf[idx] = f2bf(hsel);
}

extern "C" void kernel_launch(void* const* d_in, const int* in_sizes, int n_in,
                              void* d_out, int out_size, void* d_ws, size_t ws_size,
                              hipStream_t stream) {
    const float* features  = (const float*)d_in[0];
    const int*   captions  = (const int*)  d_in[1];
    const int*   lengths   = (const int*)  d_in[2];
    const float* temp      = (const float*)d_in[3];
    const float* embed_W   = (const float*)d_in[4];
    const float* W_enc_att = (const float*)d_in[5];
    const float* b_enc_att = (const float*)d_in[6];
    const float* W_dec_att = (const float*)d_in[7];
    const float* b_dec_att = (const float*)d_in[8];
    const float* w_full    = (const float*)d_in[9];
    const float* b_full    = (const float*)d_in[10];
    const float* W_init    = (const float*)d_in[11];
    const float* b_init    = (const float*)d_in[12];
    const float* W_beta    = (const float*)d_in[13];
    const float* b_beta    = (const float*)d_in[14];
    const float* W_ih      = (const float*)d_in[15];
    const float* b_ih      = (const float*)d_in[16];
    const float* W_hh      = (const float*)d_in[17];
    const float* b_hh      = (const float*)d_in[18];
    const float* W_out     = (const float*)d_in[19];
    const float* b_out     = (const float*)d_in[20];
    float* out = (float*)d_out;

    // ---- Workspace carve-out ----
    char* wsb = (char*)d_ws;
    auto allocF = [&](size_t n) { float* p = (float*)wsb; wsb += n * sizeof(float); return p; };
    auto allocB = [&](size_t n) { __bf16* p = (__bf16*)wsb; wsb += n * sizeof(__bf16); return p; };

    float* hc       = allocF((size_t)BB * 2 * D_DEC);
    float* enc_proj = allocF((size_t)BB * LL * D_ATT);
    float* dec_proj = allocF((size_t)BB * D_ATT);
    float* scores   = allocF((size_t)BB * LL);
    float* alpha    = allocF((size_t)BB * LL);
    float* ctx      = allocF((size_t)BB * D_ENC);
    float* gate     = allocF((size_t)BB * D_ENC);
    float* gates    = allocF((size_t)BB * 4 * D_DEC);

    __bf16* mean_bf = allocB((size_t)BB * D_ENC);
    __bf16* h_bf    = allocB((size_t)BB * D_DEC);
    __bf16* hn_bf   = allocB((size_t)BB * D_DEC);
    __bf16* x_bf    = allocB((size_t)BB * (D_EMB + D_ENC));
    __bf16* Wenc_bf = allocB((size_t)D_ATT * D_ENC);
    __bf16* Wdec_bf = allocB((size_t)D_ATT * D_DEC);
    __bf16* Winit_bf= allocB((size_t)2 * D_DEC * D_ENC);
    __bf16* Wbeta_bf= allocB((size_t)D_ENC * D_DEC);
    __bf16* Wih_bf  = allocB((size_t)4 * D_DEC * (D_EMB + D_ENC));
    __bf16* Whh_bf  = allocB((size_t)4 * D_DEC * D_DEC);
    __bf16* Wout_bf = allocB((size_t)VOCAB * D_DEC);

    // Optional 103MB bf16 features mirror (halves the dominant HBM traffic).
    const size_t featElems = (size_t)BB * LL * D_ENC;
    bool useFeatBf = ((size_t)(wsb - (char*)d_ws) + featElems * sizeof(__bf16)) <= ws_size;
    __bf16* feat_bf = useFeatBf ? allocB(featElems) : nullptr;

    auto cvt = [&](const float* in, __bf16* outp, size_t n) {
        int n4 = (int)(n / 4);
        cvt_bf16_kernel<<<(n4 + 255) / 256, 256, 0, stream>>>(in, outp, n4);
    };
    auto gemm = [&](auto A, int lda, const __bf16* W, const float* bias,
                    const float* add, int ldadd, float* outp, long long ldo,
                    int M, int N, int K, int act, const int* lens, int t) {
        int tiles = (M / 64) * (N / 16);
        int blocks = (tiles + 3) / 4;
        wmma_gemm_kernel<<<blocks, 128, 0, stream>>>(A, lda, W, bias, add, ldadd,
                                                     outp, ldo, M, N, K, act, lens, t);
    };

    // ---- One-time weight (and optionally feature) conversion ----
    cvt(W_enc_att, Wenc_bf,  (size_t)D_ATT * D_ENC);
    cvt(W_dec_att, Wdec_bf,  (size_t)D_ATT * D_DEC);
    cvt(W_init,    Winit_bf, (size_t)2 * D_DEC * D_ENC);
    cvt(W_beta,    Wbeta_bf, (size_t)D_ENC * D_DEC);
    cvt(W_ih,      Wih_bf,   (size_t)4 * D_DEC * (D_EMB + D_ENC));
    cvt(W_hh,      Whh_bf,   (size_t)4 * D_DEC * D_DEC);
    cvt(W_out,     Wout_bf,  (size_t)VOCAB * D_DEC);
    if (useFeatBf) cvt(features, feat_bf, featElems);

    // ---- Precompute ----
    mean_kernel<<<(BB * D_ENC) / 256, 256, 0, stream>>>(features, mean_bf);
    // init = mean_f @ W_init^T + b_init -> hc (B, 1024) = [h | c]
    gemm(mean_bf, D_ENC, Winit_bf, b_init, nullptr, 0, hc, 2 * D_DEC,
         BB, 2 * D_DEC, D_ENC, 0, nullptr, 0);
    hbf_init_kernel<<<(BB * D_DEC) / 256, 256, 0, stream>>>(hc, h_bf);
    // enc_proj = features @ W_enc_att^T + b_enc_att  (25088 x 512, K=2048)
    if (useFeatBf)
        gemm(feat_bf, D_ENC, Wenc_bf, b_enc_att, nullptr, 0, enc_proj, D_ATT,
             BB * LL, D_ATT, D_ENC, 0, nullptr, 0);
    else
        gemm(features, D_ENC, Wenc_bf, b_enc_att, nullptr, 0, enc_proj, D_ATT,
             BB * LL, D_ATT, D_ENC, 0, nullptr, 0);

    // ---- Decode steps ----
    for (int t = 0; t < T_STEPS; ++t) {
        // dec_proj = h @ W_dec_att^T + b
        gemm(h_bf, D_DEC, Wdec_bf, b_dec_att, nullptr, 0, dec_proj, D_ATT,
             BB, D_ATT, D_DEC, 0, nullptr, 0);
        scores_kernel<<<(BB * LL) / 8, 256, 0, stream>>>(enc_proj, dec_proj, w_full, b_full, scores);
        softmax_kernel<<<BB, 256, 0, stream>>>(scores, temp, alpha);
        if (useFeatBf)
            ctx_bf_kernel<<<(BB * D_ENC) / 256, 256, 0, stream>>>(feat_bf, alpha, ctx);
        else
            ctx_f32_kernel<<<(BB * D_ENC) / 256, 256, 0, stream>>>(features, alpha, ctx);
        // gate = sigmoid(h @ W_beta^T + b_beta)
        gemm(h_bf, D_DEC, Wbeta_bf, b_beta, nullptr, 0, gate, D_ENC,
             BB, D_ENC, D_DEC, 1, nullptr, 0);
        xasm_kernel<<<(BB * (D_EMB + D_ENC)) / 256, 256, 0, stream>>>(captions, embed_W, gate, ctx, x_bf, t);
        // gates = x @ W_ih^T + b_ih
        gemm(x_bf, D_EMB + D_ENC, Wih_bf, b_ih, nullptr, 0, gates, 4 * D_DEC,
             BB, 4 * D_DEC, D_EMB + D_ENC, 0, nullptr, 0);
        // gates += h @ W_hh^T + b_hh
        gemm(h_bf, D_DEC, Whh_bf, b_hh, gates, 4 * D_DEC, gates, 4 * D_DEC,
             BB, 4 * D_DEC, D_DEC, 0, nullptr, 0);
        lstm_kernel<<<(BB * D_DEC) / 256, 256, 0, stream>>>(gates, hc, h_bf, hn_bf, lengths, t);
        // pred = mask * (h_new @ W_out^T + b_out) -> out[:, t, :]
        gemm(hn_bf, D_DEC, Wout_bf, b_out, nullptr, 0, out + (size_t)t * VOCAB,
             (long long)T_STEPS * VOCAB, BB, VOCAB, D_DEC, 0, lengths, t);
    }
}
